// PGAT_layer_54219667145434
// MI455X (gfx1250) — matrix operand order repaired and checked by reading, not compiled, not run on hardware
//
#include <hip/hip_runtime.h>

// Problem constants (match reference)
#define NB    4
#define CLEN  1024
#define WLEN  512
#define DMODEL 512
#define NHEAD 8
#define DKH   64
#define OUTROW 2560   // 512 + 4*512

typedef __attribute__((ext_vector_type(16))) __bf16 v16bf;
typedef __attribute__((ext_vector_type(8)))  __bf16 v8bf;
typedef __attribute__((ext_vector_type(8)))  float  v8f;

union V16U { v16bf v; v8bf h[2]; };
union PackU { unsigned short us[16]; uint4 u4[2]; };

__device__ __forceinline__ unsigned short f2u_bf(float f) {
  union { float f; unsigned u; } v; v.f = f;
  unsigned r = v.u + 0x7FFFu + ((v.u >> 16) & 1u);   // round-to-nearest-even
  return (unsigned short)(r >> 16);
}
__device__ __forceinline__ unsigned pack2bf(float a, float b) {
  return (unsigned)f2u_bf(a) | ((unsigned)f2u_bf(b) << 16);
}

// A-matrix fragment (16x32 bf16): one contiguous 16-half segment per lane (k = 16*half + s).
__device__ __forceinline__ v16bf load_a_bf16(const unsigned short* rowPtr, int ks, int half) {
  V16U u;
  u.h[0] = *reinterpret_cast<const v8bf*>(rowPtr + ks + 16 * half);
  u.h[1] = *reinterpret_cast<const v8bf*>(rowPtr + ks + 16 * half + 8);
  return u.v;
}
// B-matrix fragment (32x16 bf16): one contiguous 16-half segment per lane.
__device__ __forceinline__ v16bf load_b16(const unsigned short* p) {
  return *reinterpret_cast<const v16bf*>(p);
}
__device__ __forceinline__ v8f wmma_bf16(v16bf a, v16bf b, v8f c) {
  return __builtin_amdgcn_wmma_f32_16x16x32_bf16(false, a, false, b, (short)0, c, false, false);
}
__device__ __forceinline__ v8f v8f_zero() {
  v8f z = {0.f, 0.f, 0.f, 0.f, 0.f, 0.f, 0.f, 0.f};
  return z;
}

// ---------------------------------------------------------------------------
// One-time f32 -> bf16 conversion (memory-bound; keeps RNE rounding out of
// the GEMM inner loop). One float4 per thread.
// ---------------------------------------------------------------------------
__global__ __launch_bounds__(256)
void cvt_f32_bf16(const float* __restrict__ src, unsigned* __restrict__ dst) {
  const size_t i = (size_t)blockIdx.x * 256 + threadIdx.x;
  const float4 f = reinterpret_cast<const float4*>(src)[i];
  uint2 o;
  o.x = pack2bf(f.x, f.y);
  o.y = pack2bf(f.z, f.w);
  reinterpret_cast<uint2*>(dst)[i] = o;
}

// ---------------------------------------------------------------------------
// Transpose-convert weight matrix W (512x512 f32, row-major k,n)
// into Wt (512x512 bf16, row-major n,k).
// ---------------------------------------------------------------------------
__global__ __launch_bounds__(256)
void wt_transpose(const float* __restrict__ W, unsigned short* __restrict__ Wt) {
  int idx = blockIdx.x * 256 + threadIdx.x;    // idx = n*512 + k
  int n = idx >> 9;
  int k = idx & 511;
  Wt[idx] = f2u_bf(W[(size_t)k * 512 + n]);
}

// ---------------------------------------------------------------------------
// C = Xbf(Mx512 bf16) @ Wt^T -> Out (Mx512 bf16 row-major).
// One wave computes a 64x64 tile: 256 v_wmma per wave, all operands are
// contiguous 16B/32B per-lane bf16 loads (no conversion in the loop).
// Optionally also writes per-batch transposed copy KtOut[b][d][w].
// ---------------------------------------------------------------------------
__global__ __launch_bounds__(32)
void gemm_x_wt_bf16(const unsigned short* __restrict__ Xbf,
                    const unsigned short* __restrict__ Wt,
                    unsigned short* __restrict__ Out,
                    unsigned short* __restrict__ KtOut) {
  const int lane = threadIdx.x;
  const int r = lane & 15, half = lane >> 4;
  const int m0 = blockIdx.x * 64;
  const int n0 = blockIdx.y * 64;

  v8f acc[4][4];
#pragma unroll
  for (int i = 0; i < 4; ++i)
#pragma unroll
    for (int j = 0; j < 4; ++j) acc[i][j] = v8f_zero();

  for (int k0 = 0; k0 < 512; k0 += 32) {
    v16bf A[4], Bv[4];
#pragma unroll
    for (int mi = 0; mi < 4; ++mi)
      A[mi] = load_a_bf16(Xbf + (size_t)(m0 + mi * 16 + r) * 512, k0, half);
#pragma unroll
    for (int ni = 0; ni < 4; ++ni)
      Bv[ni] = load_b16(Wt + (size_t)(n0 + ni * 16 + r) * 512 + k0 + 16 * half);
#pragma unroll
    for (int mi = 0; mi < 4; ++mi)
#pragma unroll
      for (int ni = 0; ni < 4; ++ni)
        acc[mi][ni] = wmma_bf16(A[mi], Bv[ni], acc[mi][ni]);
  }

  // D layout: N = lane%16 (column), VGPR i -> M = i + 8*(lane/16).
#pragma unroll
  for (int mi = 0; mi < 4; ++mi)
#pragma unroll
    for (int ni = 0; ni < 4; ++ni) {
      const int col = n0 + ni * 16 + r;
#pragma unroll
      for (int i = 0; i < 8; ++i) {
        const int row = m0 + mi * 16 + i + 8 * half;
        Out[(size_t)row * 512 + col] = f2u_bf(acc[mi][ni][i]);
      }
    }

  if (KtOut != nullptr) {
    const int bb = m0 >> 9;          // 64 | 512 so a tile never crosses batch
    const int wbase = m0 & 511;
#pragma unroll
    for (int mi = 0; mi < 4; ++mi)
#pragma unroll
      for (int ni = 0; ni < 4; ++ni) {
        const int d = n0 + ni * 16 + r;
        unsigned short* kp =
            KtOut + ((size_t)bb * 512 + d) * 512 + wbase + mi * 16 + 8 * half;
#pragma unroll
        for (int i = 0; i < 8; ++i) kp[i] = f2u_bf(acc[mi][ni][i]);  // contiguous
      }
  }
}

// ---------------------------------------------------------------------------
// Fused attention: one wave per (b, h, 16-row c tile).
//   QK^T (wmma) -> scores in LDS (16x512 f32), then per gate g:
//   masked softmax (f32, wave32 shuffles; contiguous 16-col slice per lane)
//   -> att bf16 in LDS, att @ K via wmma with B fragments from Kt,
//   results written straight to d_out.
// ---------------------------------------------------------------------------
__global__ __launch_bounds__(32)
void attn_fused(const unsigned short* __restrict__ Qbf,
                const unsigned short* __restrict__ Kbf,
                const unsigned short* __restrict__ Kt,
                const unsigned char* __restrict__ G,
                float* __restrict__ out) {
  __shared__ __align__(16) float s_scores[16 * 512];                // 32 KB
  __shared__ __align__(32) unsigned short s_att[16 * 512];          // 16 KB

  const int lane = threadIdx.x;
  const int r = lane & 15, half = lane >> 4;
  const int c0 = blockIdx.x * 16;
  const int h  = blockIdx.y;
  const int b  = blockIdx.z;

  const unsigned short* qrow = Qbf + (size_t)(b * CLEN + c0 + r) * 512 + h * DKH;

  // ---- QK^T : 32 w-tiles x 2 k-steps = 64 wmma ----
  for (int w0 = 0; w0 < WLEN; w0 += 16) {
    v8f acc = v8f_zero();
#pragma unroll
    for (int ks = 0; ks < DKH; ks += 32) {
      v16bf a  = load_a_bf16(qrow, ks, half);
      v16bf bm = load_b16(Kbf + (size_t)(b * WLEN + w0 + r) * 512 + h * DKH + ks + 16 * half);
      acc = wmma_bf16(a, bm, acc);
    }
#pragma unroll
    for (int i = 0; i < 8; ++i)
      s_scores[(i + 8 * half) * 512 + w0 + r] = acc[i];
  }
  __syncthreads();

  for (int g = 0; g < 4; ++g) {
    const unsigned char* gbase = G + ((size_t)(b * 4 + g) * CLEN + c0) * 512;

    // ---- masked softmax over w (scale = 1/sqrt(64) = 0.125) ----
    // Lane owns contiguous cols [lane*16, lane*16+16): G row = one b128 load,
    // scores = 4 ds_load_b128, att store = 2 packed b128 LDS stores.
    for (int row = 0; row < 16; ++row) {
      const uint4 gm = *reinterpret_cast<const uint4*>(gbase + (size_t)row * 512 + lane * 16);
      const unsigned gw[4] = {gm.x, gm.y, gm.z, gm.w};
      const float4* srow =
          reinterpret_cast<const float4*>(s_scores + row * 512 + lane * 16);
      float sarr[16];
#pragma unroll
      for (int q = 0; q < 4; ++q) {
        const float4 sf = srow[q];
        sarr[q * 4 + 0] = sf.x; sarr[q * 4 + 1] = sf.y;
        sarr[q * 4 + 2] = sf.z; sarr[q * 4 + 3] = sf.w;
      }
      unsigned maskbits = 0;
      float mx = -3.0e38f;
#pragma unroll
      for (int j = 0; j < 16; ++j) {
        const unsigned m = (gw[j >> 2] >> ((j & 3) * 8)) & 0xFFu;
        const float s = m ? sarr[j] * 0.125f : -1.0e9f;
        maskbits |= (m ? 1u : 0u) << j;
        sarr[j] = s;
        mx = fmaxf(mx, s);
      }
#pragma unroll
      for (int off = 16; off >= 1; off >>= 1) mx = fmaxf(mx, __shfl_xor(mx, off, 32));
      float sum = 0.f;
#pragma unroll
      for (int j = 0; j < 16; ++j) { const float p = __expf(sarr[j] - mx); sarr[j] = p; sum += p; }
#pragma unroll
      for (int off = 16; off >= 1; off >>= 1) sum += __shfl_xor(sum, off, 32);
      const float inv = 1.0f / sum;
      PackU pk;
#pragma unroll
      for (int j = 0; j < 16; ++j) {
        const float a = ((maskbits >> j) & 1u) ? sarr[j] * inv : 0.0f;
        pk.us[j] = f2u_bf(a);
      }
      uint4* adst = reinterpret_cast<uint4*>(s_att + row * 512 + lane * 16);
      adst[0] = pk.u4[0];
      adst[1] = pk.u4[1];
    }
    __syncthreads();

    // ---- att @ K : 4 n-tiles x 16 k-steps = 64 wmma per gate ----
#pragma unroll
    for (int nt = 0; nt < 4; ++nt) {
      v8f acc = v8f_zero();
      for (int w0 = 0; w0 < WLEN; w0 += 32) {
        v16bf a  = load_a_bf16(s_att + (size_t)r * 512, w0, half);
        v16bf bm = load_b16(Kt + ((size_t)b * 512 + h * DKH + nt * 16 + r) * 512 + w0 + 16 * half);
        acc = wmma_bf16(a, bm, acc);
      }
      const int col = 512 + g * 512 + h * DKH + nt * 16 + r;
#pragma unroll
      for (int i = 0; i < 8; ++i) {
        const int c = c0 + i + 8 * half;
        out[(size_t)(b * CLEN + c) * OUTROW + col] = acc[i];
      }
    }
    __syncthreads();
  }
}

// ---------------------------------------------------------------------------
// Pass-through: out[b,c,0:512] = chars[b,c,:]
// ---------------------------------------------------------------------------
__global__ __launch_bounds__(128)
void copy_chars(const float* __restrict__ chars, float* __restrict__ out) {
  const int row = blockIdx.x;            // 0 .. 4095 (b*1024+c)
  const int t = threadIdx.x;             // 128 threads x float4 = 512 floats
  const float4* src = reinterpret_cast<const float4*>(chars + (size_t)row * 512);
  float4* dst = reinterpret_cast<float4*>(out + (size_t)row * OUTROW);
  dst[t] = src[t];
}

extern "C" void kernel_launch(void* const* d_in, const int* in_sizes, int n_in,
                              void* d_out, int out_size, void* d_ws, size_t ws_size,
                              hipStream_t stream) {
  (void)in_sizes; (void)n_in; (void)out_size; (void)ws_size;
  const float* chars        = (const float*)d_in[0];
  const float* words        = (const float*)d_in[1];
  const unsigned char* G    = (const unsigned char*)d_in[2];  // numpy bool = 1 byte
  const float* W_Q          = (const float*)d_in[3];
  const float* W_K          = (const float*)d_in[4];
  float* out                = (float*)d_out;

  char* ws = (char*)d_ws;
  const size_t SZ_W   = (size_t)512 * 512 * 2;        // 0.5 MB bf16 weight
  const size_t SZ_Q   = (size_t)4096 * 512 * 2;       // 4 MB (Q / chars-bf16)
  const size_t SZ_K   = (size_t)2048 * 512 * 2;       // 2 MB (K / words-bf16)
  unsigned short* WtQ  = (unsigned short*)(ws);
  unsigned short* WtK  = (unsigned short*)(ws + SZ_W);
  unsigned short* Qbf  = (unsigned short*)(ws + 2 * SZ_W);
  unsigned short* Kbf  = (unsigned short*)(ws + 2 * SZ_W + SZ_Q);
  unsigned short* Kt   = (unsigned short*)(ws + 2 * SZ_W + SZ_Q + SZ_K);
  unsigned short* Cbf  = (unsigned short*)(ws + 2 * SZ_W + SZ_Q + 2 * SZ_K);
  unsigned short* Wdbf = (unsigned short*)(ws + 2 * SZ_W + 2 * SZ_Q + 2 * SZ_K);
  // total workspace: ~15 MB

  // One-time input conversion to bf16 (memory-bound)
  cvt_f32_bf16<<<dim3((4096 * 512 / 4) / 256), 256, 0, stream>>>(chars, (unsigned*)Cbf);
  cvt_f32_bf16<<<dim3((2048 * 512 / 4) / 256), 256, 0, stream>>>(words, (unsigned*)Wdbf);
  wt_transpose<<<dim3((512 * 512) / 256), 256, 0, stream>>>(W_Q, WtQ);
  wt_transpose<<<dim3((512 * 512) / 256), 256, 0, stream>>>(W_K, WtK);

  // Q = chars @ W_Q  (M = 4096) ; K = words @ W_K (M = 2048, + transposed Kt)
  gemm_x_wt_bf16<<<dim3(64, 8), 32, 0, stream>>>(Cbf, WtQ, Qbf, nullptr);
  gemm_x_wt_bf16<<<dim3(32, 8), 32, 0, stream>>>(Wdbf, WtK, Kbf, Kt);

  // Fused masked-softmax attention over (b, h, c-tile)
  attn_fused<<<dim3(CLEN / 16, NHEAD, NB), 32, 0, stream>>>(Qbf, Kbf, Kt, G, out);

  // chars pass-through into first 512 output columns
  copy_chars<<<dim3(NB * CLEN), 128, 0, stream>>>(chars, out);
}
